// SparseCIN_80298708566557
// MI455X (gfx1250) — matrix operand order repaired and checked by reading, not compiled, not run on hardware
//
#include <hip/hip_runtime.h>

// ---------------------------------------------------------------------------
// SparseCIN forward for MI455X (gfx1250, wave32, WMMA).
// Dense GEMMs run on v_wmma_f32_16x16x32_bf16 (fp32 accumulate, bf16 operands
// built in registers from fp32 storage). BatchNorm+ReLU is folded into the
// A-operand load of the consuming GEMM; each GEMM emits column sum/sumsq for
// the *next* BN via LDS-reduced atomics. All pre-BN linear biases cancel.
// Block = 256 threads (8 waves): 4 row-waves x 2 column-groups, 4 accum tiles
// per wave (<256 VGPRs, no s_set_vgpr_msb). bf16 pair-pack via v_perm_b32.
// ---------------------------------------------------------------------------

typedef __attribute__((ext_vector_type(16))) __bf16 v16bf;
typedef __attribute__((ext_vector_type(8)))  float  v8f;

union BFrag { v16bf v; unsigned u[8]; };
union CFrag { v8f   v; float    f[8]; };

// Pack two f32 into packed bf16 {hi16(b),hi16(a)} with a single v_perm_b32.
// (Truncating round; every GEMM output is BatchNorm-renormalized downstream.)
__device__ __forceinline__ unsigned pack2bf(float lo, float hi) {
  union { float f; unsigned u; } a, b;
  a.f = lo; b.f = hi;
  return __builtin_amdgcn_perm(b.u, a.u, 0x07060302u); // {b.b3,b.b2,a.b3,a.b2}
}

// flags: 1 = apply relu(z*s+t) to A while loading; 2 = accumulate into Z;
//        4 = emit column sum / sumsq stats.
__global__ void __launch_bounds__(256)
gemm128_bn_wmma(float* __restrict__ Z, const float* __restrict__ A,
                const float* __restrict__ W,                 // [128][128] row-major (K x Nout)
                const float* __restrict__ sIn, const float* __restrict__ tIn,
                float* __restrict__ colsum, float* __restrict__ colsq,
                int N, int flags)
{
  __shared__ unsigned lds_w[8192];                       // 4 ksteps x 8 tiles x 32 lanes x 8 dwords
  __shared__ float lds_s[128], lds_t[128], lds_cs[128], lds_cq[128];

  const int tid = threadIdx.x;

  // Stage W into LDS pre-swizzled to the exact per-lane B-fragment layout:
  // frag(ks,tile,lane): n = tile*16 + (lane&15), k = ks*32 + (lane>>4)*8 + 2v{,+1} (+8 for v>=4)
  for (int idx = tid; idx < 1024; idx += 256) {
    int ks = idx >> 8, rem = idx & 255, tile = rem >> 5, ln = rem & 31;
    int n  = tile * 16 + (ln & 15);
    int k0 = ks * 32 + (ln >> 4) * 8;
    unsigned* dst = lds_w + idx * 8;
#pragma unroll
    for (int j = 0; j < 8; ++j)
      dst[j] = pack2bf(W[(k0 + 2 * j + (j >> 2) * 8) * 128 + n], W[(k0 + 2 * j + (j >> 2) * 8 + 1) * 128 + n]);
  }
  if (tid < 128) {
    lds_s[tid]  = (flags & 1) ? sIn[tid] : 1.0f;
    lds_t[tid]  = (flags & 1) ? tIn[tid] : 0.0f;
    lds_cs[tid] = 0.0f;
    lds_cq[tid] = 0.0f;
  }
  __syncthreads();

  const int warp = tid >> 5, lane = tid & 31;
  const int rwave = warp & 3;                            // row-wave 0..3
  const int cgrp  = warp >> 2;                           // column group 0..1
  const int half  = lane >> 4, lr = lane & 15;
  const bool act = (flags & 1), accum = (flags & 2), stats = (flags & 4);

  for (int rb = blockIdx.x * 64; rb < N; rb += gridDim.x * 64) {
    const int rowbase = rb + rwave * 16;
    int row  = rowbase + lr;
    int rowc = row < N ? row : N - 1;                    // clamp: EXEC stays all-ones
    const float* Ar = A + (size_t)rowc * 128;

    CFrag acc[4];
#pragma unroll
    for (int t = 0; t < 4; ++t)
#pragma unroll
      for (int r = 0; r < 8; ++r) acc[t].f[r] = 0.0f;

#pragma unroll
    for (int ks = 0; ks < 4; ++ks) {
      // A fragment: lane holds row (lane&15), K = ks*32 + half*8 + {0..7, 16..23}
      const float* ap = Ar + ks * 32 + half * 8;
      float f[16];
      float4 q;
      q = *(const float4*)(ap +  0); f[0]  = q.x; f[1]  = q.y; f[2]  = q.z; f[3]  = q.w;
      q = *(const float4*)(ap +  4); f[4]  = q.x; f[5]  = q.y; f[6]  = q.z; f[7]  = q.w;
      q = *(const float4*)(ap + 16); f[8]  = q.x; f[9]  = q.y; f[10] = q.z; f[11] = q.w;
      q = *(const float4*)(ap + 20); f[12] = q.x; f[13] = q.y; f[14] = q.z; f[15] = q.w;
      if (act) {
        int cb = ks * 32 + half * 8;
#pragma unroll
        for (int j = 0; j < 8; ++j)
          f[j]     = fmaxf(f[j]     * lds_s[cb + j]      + lds_t[cb + j],      0.0f);
#pragma unroll
        for (int j = 0; j < 8; ++j)
          f[8 + j] = fmaxf(f[8 + j] * lds_s[cb + 16 + j] + lds_t[cb + 16 + j], 0.0f);
      }
      BFrag a;
#pragma unroll
      for (int v = 0; v < 8; ++v) a.u[v] = pack2bf(f[2 * v], f[2 * v + 1]);

#pragma unroll
      for (int t = 0; t < 4; ++t) {
        const int tg = cgrp * 4 + t;                     // global column tile
        const uint4* bp = (const uint4*)(lds_w + ((ks * 8 + tg) * 32 + lane) * 8);
        uint4 b0 = bp[0], b1 = bp[1];
        BFrag b;
        b.u[0] = b0.x; b.u[1] = b0.y; b.u[2] = b0.z; b.u[3] = b0.w;
        b.u[4] = b1.x; b.u[5] = b1.y; b.u[6] = b1.z; b.u[7] = b1.w;
        acc[t].v = __builtin_amdgcn_wmma_f32_16x16x32_bf16(
            false, a.v, false, b.v, (short)0, acc[t].v, false, false);
      }
    }

    // Writeback (+ optional accumulate) and per-column stats.
#pragma unroll
    for (int t = 0; t < 4; ++t) {
      int col = (cgrp * 4 + t) * 16 + lr;
      float psum = 0.0f, psq = 0.0f;
#pragma unroll
      for (int r = 0; r < 8; ++r) {
        int rr = rowbase + r + 8 * half;                 // C layout: VGPR r -> M=r (+8 for hi half)
        if (rr < N) {
          size_t off = (size_t)rr * 128 + col;
          float v = acc[t].f[r];
          if (accum) v += Z[off];
          Z[off] = v;
          psum += v; psq += v * v;
        }
      }
      if (stats) { atomicAdd(&lds_cs[col], psum); atomicAdd(&lds_cq[col], psq); }
    }
  }
  __syncthreads();
  if (stats && tid < 128) {
    atomicAdd(&colsum[tid], lds_cs[tid]);
    atomicAdd(&colsq[tid],  lds_cq[tid]);
  }
}

// Fold batch stats + gamma/beta into per-column scale/shift: bn(z)=z*s+t.
__global__ void bn_finalize(float* __restrict__ s, float* __restrict__ t,
                            const float* __restrict__ cs, const float* __restrict__ cq,
                            const float* __restrict__ g, const float* __restrict__ be,
                            float invN)
{
  int c = threadIdx.x;
  float m  = cs[c] * invN;
  float v  = cq[c] * invN - m * m;
  float sc = g[c] * rsqrtf(v + 1e-5f);
  s[c] = sc;
  t[c] = be[c] - m * sc;
}

// y[dst[e]] += x[src[e]]  (one wave per edge row, float4 per lane)
__global__ void scatter_add(float* __restrict__ y, const float* __restrict__ x,
                            const int* __restrict__ src, const int* __restrict__ dst, int E)
{
  int tgl = blockIdx.x * blockDim.x + threadIdx.x;
  int e = tgl >> 5;
  if (e >= E) return;
  int l = tgl & 31;
  const float4 v = *(const float4*)(x + (size_t)src[e] * 128 + l * 4);
  float* yp = y + (size_t)dst[e] * 128 + l * 4;
  atomicAdd(yp + 0, v.x); atomicAdd(yp + 1, v.y);
  atomicAdd(yp + 2, v.z); atomicAdd(yp + 3, v.w);
}

// pooled[batch[i]] += x[i]
__global__ void pool_scatter(float* __restrict__ pooled, const float* __restrict__ x,
                             const int* __restrict__ batch, int Nn)
{
  int tgl = blockIdx.x * blockDim.x + threadIdx.x;
  int i = tgl >> 5;
  if (i >= Nn) return;
  int l = tgl & 31;
  const float4 v = *(const float4*)(x + (size_t)i * 128 + l * 4);
  float* p = pooled + (size_t)batch[i] * 128 + l * 4;
  atomicAdd(p + 0, v.x); atomicAdd(p + 1, v.y);
  atomicAdd(p + 2, v.z); atomicAdd(p + 3, v.w);
}

__global__ void bnrelu_inplace(float* __restrict__ x, const float* __restrict__ s,
                               const float* __restrict__ t, long long total)
{
  long long i = (long long)blockIdx.x * blockDim.x + threadIdx.x;
  if (i >= total) return;
  int c = (int)(i & 127);
  x[i] = fmaxf(x[i] * s[c] + t[c], 0.0f);
}

// out[b][n] = sum_d relu(bias_d[n] + pooled_d[b] . W_d[:,n])
__global__ void readout_kernel(float* __restrict__ out, const float* __restrict__ pooled,
                               const float* __restrict__ W0, const float* __restrict__ b0,
                               const float* __restrict__ W1, const float* __restrict__ b1,
                               const float* __restrict__ W2, const float* __restrict__ b2,
                               int B)
{
  int idx = blockIdx.x * blockDim.x + threadIdx.x;
  if (idx >= B * 256) return;
  int b = idx >> 8, n = idx & 255;
  const float* Ws[3] = {W0, W1, W2};
  const float* bs[3] = {b0, b1, b2};
  float acc = 0.0f;
  for (int d = 0; d < 3; ++d) {
    const float* p = pooled + ((size_t)d * B + b) * 128;
    float s = bs[d][n];
#pragma unroll 8
    for (int k = 0; k < 128; ++k) s += p[k] * Ws[d][k * 256 + n];
    acc += fmaxf(s, 0.0f);
  }
  out[idx] = acc;
}

extern "C" void kernel_launch(void* const* d_in, const int* in_sizes, int n_in,
                              void* d_out, int out_size, void* d_ws, size_t ws_size,
                              hipStream_t stream)
{
  const int H = 128;
  const float* xin[3] = {(const float*)d_in[0], (const float*)d_in[1], (const float*)d_in[2]};
  int N[3] = {in_sizes[0] / H, in_sizes[1] / H, in_sizes[2] / H};
  const int* upi[2]  = {(const int*)d_in[3], (const int*)d_in[4]};
  int EU[2] = {in_sizes[3] / 2, in_sizes[4] / 2};
  const int* bndi[2] = {(const int*)d_in[5], (const int*)d_in[6]};
  int EB[2] = {in_sizes[5] / 2, in_sizes[6] / 2};
  const int* batch[3] = {(const int*)d_in[7], (const int*)d_in[8], (const int*)d_in[9]};
  const int Bsz = out_size / 256;

  // params flattened (jax dict keys sorted): per (layer,dim) 20 leaves:
  // bd.{W1,W2,b1,b2,be1,be2,g1,g2}, cb.{W,b,be,g}, up.{W1,W2,b1,b2,be1,be2,g1,g2};
  // then lin1: 3 x {W,b}. Biases before BN cancel -> unused.
  auto P = [&](int i) { return (const float*)d_in[10 + i]; };

  // ---- workspace carve ----
  char* w = (char*)d_ws;
  auto carve = [&](size_t bytes) { void* p = (void*)w; w += (bytes + 255) & ~(size_t)255; return p; };
  float* xA[3]; float* xB[3];
  for (int d = 0; d < 3; ++d) xA[d] = (float*)carve((size_t)N[d] * H * 4);
  for (int d = 0; d < 3; ++d) xB[d] = (float*)carve((size_t)N[d] * H * 4);
  size_t Nmax = (size_t)(N[0] > N[1] ? N[0] : N[1]); if ((size_t)N[2] > Nmax) Nmax = N[2];
  float* S0 = (float*)carve(Nmax * H * 4);   // y = x + agg
  float* S1 = (float*)carve(Nmax * H * 4);   // z1
  float* S2 = (float*)carve(Nmax * H * 4);   // z2 (up)
  float* S3 = (float*)carve(Nmax * H * 4);   // z2 (bd)
  float* stat   = (float*)carve(2048 * 4);
  float* pooled = (float*)carve((size_t)3 * Bsz * H * 4);

  float* cs  = stat,        *cq  = stat + 128;
  float* s1  = stat + 256,  *t1  = stat + 384;
  float* s2u = stat + 512,  *t2u = stat + 640;
  float* s2b = stat + 768,  *t2b = stat + 896;
  float* sc  = stat + 1024, *tc  = stat + 1152;

  auto gemm = [&](float* Z, const float* A, const float* W,
                  const float* si, const float* ti, int n, int flags) {
    int nb = (n + 63) / 64; if (nb > 4096) nb = 4096; if (nb < 1) nb = 1;
    hipLaunchKernelGGL(gemm128_bn_wmma, dim3(nb), dim3(256), 0, stream,
                       Z, A, W, si, ti, cs, cq, n, flags);
  };
  auto zstats = [&]() { hipMemsetAsync(cs, 0, 256 * 4, stream); };
  auto fin = [&](float* so, float* to, const float* g, const float* be, int n) {
    hipLaunchKernelGGL(bn_finalize, dim3(1), dim3(128), 0, stream,
                       so, to, cs, cq, g, be, 1.0f / (float)n);
  };

  for (int L = 0; L < 3; ++L) {
    const float* xc[3]; float* xo[3];
    for (int d = 0; d < 3; ++d) {
      if (L == 0)      { xc[d] = xin[d]; xo[d] = xA[d]; }
      else if (L == 1) { xc[d] = xA[d];  xo[d] = xB[d]; }
      else             { xc[d] = xB[d];  xo[d] = xA[d]; }
    }
    for (int d = 0; d < 3; ++d) {
      int base = (L * 3 + d) * 20;
      const float *bdW1 = P(base + 0),  *bdW2 = P(base + 1);
      const float *bdbe1 = P(base + 4), *bdbe2 = P(base + 5);
      const float *bdg1 = P(base + 6),  *bdg2 = P(base + 7);
      const float *cbW = P(base + 8),   *cbbe = P(base + 10), *cbg = P(base + 11);
      const float *upW1 = P(base + 12), *upW2 = P(base + 13);
      const float *upbe1 = P(base + 16), *upbe2 = P(base + 17);
      const float *upg1 = P(base + 18), *upg2 = P(base + 19);
      const size_t xbytes = (size_t)N[d] * H * 4;

      // ---- up path: y = x + segsum(x[src] -> dst) ----
      const float* Ay;
      if (d < 2) {
        hipMemcpyAsync(S0, xc[d], xbytes, hipMemcpyDeviceToDevice, stream);
        int nb = (EU[d] * 32 + 255) / 256;
        hipLaunchKernelGGL(scatter_add, dim3(nb), dim3(256), 0, stream,
                           S0, xc[d], upi[d], upi[d] + EU[d], EU[d]);
        Ay = S0;
      } else Ay = xc[2];
      zstats(); gemm(S1, Ay, upW1, nullptr, nullptr, N[d], 4);  fin(s1, t1, upg1, upbe1, N[d]);
      zstats(); gemm(S2, S1, upW2, s1, t1, N[d], 1 | 4);        fin(s2u, t2u, upg2, upbe2, N[d]);

      // ---- bd path: y = x + segsum(x_{d-1}[src] -> dst) ----
      if (d > 0) {
        hipMemcpyAsync(S0, xc[d], xbytes, hipMemcpyDeviceToDevice, stream);
        int e = EB[d - 1];
        int nb = (e * 32 + 255) / 256;
        hipLaunchKernelGGL(scatter_add, dim3(nb), dim3(256), 0, stream,
                           S0, xc[d - 1], bndi[d - 1], bndi[d - 1] + e, e);
        Ay = S0;
      } else Ay = xc[0];
      zstats(); gemm(S1, Ay, bdW1, nullptr, nullptr, N[d], 4);  fin(s1, t1, bdg1, bdbe1, N[d]);
      zstats(); gemm(S3, S1, bdW2, s1, t1, N[d], 1 | 4);        fin(s2b, t2b, bdg2, bdbe2, N[d]);

      // ---- combine: [relu(bn(z2u)) | relu(bn(z2b))] @ Wc, as two K=128 passes ----
      gemm(xo[d], S2, cbW, s2u, t2u, N[d], 1);
      zstats(); gemm(xo[d], S3, cbW + 128 * 128, s2b, t2b, N[d], 1 | 2 | 4);
      fin(sc, tc, cbg, cbbe, N[d]);
      {
        long long tot = (long long)N[d] * H;
        int nb = (int)((tot + 255) / 256);
        hipLaunchKernelGGL(bnrelu_inplace, dim3(nb), dim3(256), 0, stream,
                           xo[d], sc, tc, tot);
      }
    }
  }

  // ---- readout: per-dim batch pooling, lin1 + relu, sum over dims ----
  hipMemsetAsync(pooled, 0, (size_t)3 * Bsz * H * 4, stream);
  for (int d = 0; d < 3; ++d) {
    int nb = (N[d] * 32 + 255) / 256;
    hipLaunchKernelGGL(pool_scatter, dim3(nb), dim3(256), 0, stream,
                       pooled + (size_t)d * Bsz * H, xA[d], batch[d], N[d]);
  }
  {
    int nb = (Bsz * 256 + 255) / 256;
    hipLaunchKernelGGL(readout_kernel, dim3(nb), dim3(256), 0, stream,
                       (float*)d_out, pooled,
                       P(180), P(181), P(182), P(183), P(184), P(185), Bsz);
  }
  (void)n_in; (void)ws_size;
}
